// ALIGNNConvPyG_10831907521230
// MI455X (gfx1250) — matrix operand order repaired and checked by reading, not compiled, not run on hardware
//
#include <hip/hip_runtime.h>
#include <hip/hip_bf16.h>

// ---------------------------------------------------------------------------
// ALIGNN conv (two EdgeGatedGraphConv passes) for MI455X / gfx1250.
// GEMMs in bf16 via v_wmma_f32_16x16x32_bf16 (wave32, 16x16 tiles, K=128).
// The 4 node projections (sg/dg/du/su) are fused into one kernel: each A
// fragment is loaded once and reused across 4 weight sets (4x arithmetic
// intensity, 16 WMMAs per wave-tile). Gather/scatter + segment-sum via global
// f32 atomics (memory-bound at 23.3 TB/s).
// ---------------------------------------------------------------------------

#define H 128
static const long long Nn   = 50000;    // atoms
static const long long EGn  = 600000;   // bonds  (nodes of line graph)
static const long long ELGn = 1200000;  // triplets (edges of line graph)

typedef __attribute__((ext_vector_type(16))) __bf16 v16bf;
typedef __attribute__((ext_vector_type(8)))  float  v8f;

__device__ __forceinline__ float silu_f(float g) {
    return g / (1.0f + __expf(-g));
}

// ---------------------------- elementwise utils ----------------------------

__global__ void cvt_f32_to_bf16_kernel(const float* __restrict__ in,
                                       __bf16* __restrict__ out, long long n) {
    long long i = (long long)blockIdx.x * blockDim.x + threadIdx.x;
    if (i < n) out[i] = (__bf16)in[i];
}

__global__ void fill_f32_kernel(float* __restrict__ p, float v, long long n) {
    long long i = (long long)blockIdx.x * blockDim.x + threadIdx.x;
    if (i < n) p[i] = v;
}

// ------------------------- WMMA fragment helpers ---------------------------
// Fragment layouts per CDNA5 ISA §7.12.2 (wave32):
//   A 16x32 bf16 : lane = {row r = lane&15, half h = lane>>4}; vgpr pair j ->
//                  K = (j>>2)*16 + h*8 + (j&3)*2 + {0,1}
//   B 32x16 bf16 : lane col n = lane&15; same K mapping as A (h = lane>>4)
//   D 16x16 f32  : lane col n = lane&15; vgpr v -> row M = (lane>>4)*8 + v

__device__ __forceinline__ v16bf load_a_frag(const __bf16* __restrict__ arow,
                                             int k0, int hs) {
    v16bf a;
#pragma unroll
    for (int j = 0; j < 8; ++j) {
        const int ka = k0 + (j >> 2) * 16 + hs * 8 + (j & 3) * 2;
        unsigned ua = *(const unsigned*)(arow + ka);
        a[2 * j]     = ((const __bf16*)&ua)[0];
        a[2 * j + 1] = ((const __bf16*)&ua)[1];
    }
    return a;
}

__device__ __forceinline__ v16bf load_b_frag(const __bf16* __restrict__ wrow,
                                             int k0, int hs) {
    v16bf b;
#pragma unroll
    for (int j = 0; j < 8; ++j) {
        const int kb = k0 + (j >> 2) * 16 + hs * 8 + (j & 3) * 2;
        unsigned ub = *(const unsigned*)(wrow + kb);
        b[2 * j]     = ((const __bf16*)&ub)[0];
        b[2 * j + 1] = ((const __bf16*)&ub)[1];
    }
    return b;
}

// ----------------------- single-output WMMA GEMM ---------------------------
// C[M,128] = A[M,128] @ W^T + bias (used for the edge projection Weg).
__global__ void gemm128_bf16_wmma_kernel(const __bf16* __restrict__ A,
                                         const __bf16* __restrict__ W,
                                         const float* __restrict__ bias,
                                         float* __restrict__ C, int M) {
    const int lane  = threadIdx.x & 31;
    const int ntile = threadIdx.x >> 5;      // 0..7
    const int r     = lane & 15;
    const int hs    = lane >> 4;
    const int mbase = blockIdx.x * 16;
    const int n     = ntile * 16 + r;

    const __bf16* __restrict__ arow = A + (size_t)(mbase + r) * H;
    const __bf16* __restrict__ wrow = W + (size_t)n * H;

    v8f acc = {};
#pragma unroll
    for (int k0 = 0; k0 < H; k0 += 32) {
        v16bf a = load_a_frag(arow, k0, hs);
        v16bf b = load_b_frag(wrow, k0, hs);
        acc = __builtin_amdgcn_wmma_f32_16x16x32_bf16(
            false, a, false, b, (short)0, acc, false, false);
    }

    const float bv = bias[n];
#pragma unroll
    for (int v = 0; v < 8; ++v) {
        const int m = mbase + hs * 8 + v;
        C[(size_t)m * H + n] = acc[v] + bv;
    }
}

// ------------------- fused 4-output WMMA GEMM (node proj) ------------------
// C_t[M,128] = A[M,128] @ W_t^T + b_t for t = 0..3 (sg, dg, du, su).
// A fragments loaded once per k-step and reused by 4 WMMA chains.
// W4 = 4 consecutive [128,128] bf16 weight matrices.
__global__ void gemm128x4_bf16_wmma_kernel(const __bf16* __restrict__ A,
                                           const __bf16* __restrict__ W4,
                                           const float* __restrict__ b0,
                                           const float* __restrict__ b1,
                                           const float* __restrict__ b2,
                                           const float* __restrict__ b3,
                                           float* __restrict__ C0,
                                           float* __restrict__ C1,
                                           float* __restrict__ C2,
                                           float* __restrict__ C3, int M) {
    const int lane  = threadIdx.x & 31;
    const int ntile = threadIdx.x >> 5;      // 0..7
    const int r     = lane & 15;
    const int hs    = lane >> 4;
    const int mbase = blockIdx.x * 16;
    const int n     = ntile * 16 + r;
    const long long HH = (long long)H * H;

    const __bf16* __restrict__ arow = A + (size_t)(mbase + r) * H;
    const __bf16* __restrict__ w0 = W4 + 0 * HH + (size_t)n * H;
    const __bf16* __restrict__ w1 = W4 + 1 * HH + (size_t)n * H;
    const __bf16* __restrict__ w2 = W4 + 2 * HH + (size_t)n * H;
    const __bf16* __restrict__ w3 = W4 + 3 * HH + (size_t)n * H;

    v8f acc0 = {}, acc1 = {}, acc2 = {}, acc3 = {};
#pragma unroll
    for (int k0 = 0; k0 < H; k0 += 32) {
        const v16bf a = load_a_frag(arow, k0, hs);   // loaded once, used 4x
        v16bf bb;
        bb = load_b_frag(w0, k0, hs);
        acc0 = __builtin_amdgcn_wmma_f32_16x16x32_bf16(false, a, false, bb, (short)0, acc0, false, false);
        bb = load_b_frag(w1, k0, hs);
        acc1 = __builtin_amdgcn_wmma_f32_16x16x32_bf16(false, a, false, bb, (short)0, acc1, false, false);
        bb = load_b_frag(w2, k0, hs);
        acc2 = __builtin_amdgcn_wmma_f32_16x16x32_bf16(false, a, false, bb, (short)0, acc2, false, false);
        bb = load_b_frag(w3, k0, hs);
        acc3 = __builtin_amdgcn_wmma_f32_16x16x32_bf16(false, a, false, bb, (short)0, acc3, false, false);
    }

    const float v0 = b0[n], v1 = b1[n], v2 = b2[n], v3 = b3[n];
#pragma unroll
    for (int v = 0; v < 8; ++v) {
        const size_t m = (size_t)(mbase + hs * 8 + v) * H + n;
        C0[m] = acc0[v] + v0;
        C1[m] = acc1[v] + v1;
        C2[m] = acc2[v] + v2;
        C3[m] = acc3[v] + v3;
    }
}

// ------------------------------ edge kernels -------------------------------
// gate = sg[dst] + dg[src] + eg_lin ; sigma = silu(gate)
// pass 1: norm[dst] += sigma          (norm pre-filled with 1e-8)
// pass 2: out[dst]  += sigma * du[src] / norm[dst]   (out pre-filled with su)

__global__ void edge_norm_kernel(const int* __restrict__ src,
                                 const int* __restrict__ dst,
                                 const float* __restrict__ sg,
                                 const float* __restrict__ dg,
                                 const float* __restrict__ eglin,
                                 float* __restrict__ norm, long long EH) {
    long long idx = (long long)blockIdx.x * blockDim.x + threadIdx.x;
    if (idx >= EH) return;
    const long long e = idx >> 7;
    const int c = (int)(idx & (H - 1));
    const int s = src[e], d = dst[e];
    const float g = sg[(size_t)d * H + c] + dg[(size_t)s * H + c] + eglin[idx];
    atomicAdd(norm + (size_t)d * H + c, silu_f(g));
}

__global__ void edge_msg_kernel(const int* __restrict__ src,
                                const int* __restrict__ dst,
                                const float* __restrict__ sg,
                                const float* __restrict__ dg,
                                const float* __restrict__ eglin,
                                const float* __restrict__ du,
                                const float* __restrict__ norm,
                                float* __restrict__ out, long long EH) {
    long long idx = (long long)blockIdx.x * blockDim.x + threadIdx.x;
    if (idx >= EH) return;
    const long long e = idx >> 7;
    const int c = (int)(idx & (H - 1));
    const int s = src[e], d = dst[e];
    const float g = sg[(size_t)d * H + c] + dg[(size_t)s * H + c] + eglin[idx];
    const float sig = silu_f(g);
    const float m = sig * du[(size_t)s * H + c] / norm[(size_t)d * H + c];
    atomicAdd(out + (size_t)d * H + c, m);
}

// ------------------------------- batch norm --------------------------------

__global__ void bn_stats_kernel(const float* __restrict__ out,
                                float* __restrict__ stats, long long V) {
    __shared__ float sh[256];
    const int c = threadIdx.x & (H - 1);
    const int sub = threadIdx.x >> 7;  // 0..1
    float s = 0.f, sq = 0.f;
    for (long long r = (long long)blockIdx.x * 2 + sub; r < V;
         r += (long long)gridDim.x * 2) {
        const float v = out[r * H + c];
        s += v;
        sq += v * v;
    }
    sh[threadIdx.x] = s;
    __syncthreads();
    if (sub == 0) s += sh[128 + c];
    __syncthreads();
    sh[threadIdx.x] = sq;
    __syncthreads();
    if (sub == 0) {
        sq += sh[128 + c];
        atomicAdd(&stats[c], s);
        atomicAdd(&stats[128 + c], sq);
    }
}

__global__ void bn_finalize_kernel(float* __restrict__ stats,
                                   const float* __restrict__ gamma,
                                   const float* __restrict__ beta, float V) {
    const int c = threadIdx.x;
    if (c < H) {
        const float mean = stats[c] / V;
        const float var = stats[128 + c] / V - mean * mean;
        const float scale = gamma[c] * rsqrtf(var + 1e-5f);
        stats[c] = scale;
        stats[128 + c] = beta[c] - mean * scale;
    }
}

// out = silu(out*scale + shift) + x ; optional bf16 copy for the next conv.
__global__ void bn_apply_kernel(float* __restrict__ out,
                                const float* __restrict__ x,
                                const float* __restrict__ stats,
                                __bf16* __restrict__ bf_out, long long VH) {
    long long idx = (long long)blockIdx.x * blockDim.x + threadIdx.x;
    if (idx >= VH) return;
    const int c = (int)(idx & (H - 1));
    float o = out[idx] * stats[c] + stats[128 + c];
    o = silu_f(o) + x[idx];
    out[idx] = o;
    if (bf_out) bf_out[idx] = (__bf16)o;
}

// ----------------------------- orchestration -------------------------------

struct ConvParams {
    const float *Wsg, *bsg, *Wdg, *bdg, *Weg, *beg, *Wsu, *bsu, *Wdu, *bdu;
    const float *gamma, *beta;
};

static inline unsigned nblk(long long n, int b) {
    return (unsigned)((n + b - 1) / b);
}

static void run_conv(hipStream_t stream, const float* x, long long V,
                     const int* src, const int* dst, long long E,
                     const __bf16* xb, const __bf16* eab, __bf16* wb,
                     float* sg, float* dg, float* du, float* eglin,
                     float* norm, float* stats, const ConvParams& P,
                     float* out, __bf16* out_bf16) {
    const long long HH = (long long)H * H;
    const long long VH = V * H;
    const long long EH = E * H;

    // bf16 weight copies: slots 0=Wsg 1=Wdg 2=Wdu 3=Wsu (fused GEMM) 4=Weg
    cvt_f32_to_bf16_kernel<<<nblk(HH, 256), 256, 0, stream>>>(P.Wsg, wb + 0 * HH, HH);
    cvt_f32_to_bf16_kernel<<<nblk(HH, 256), 256, 0, stream>>>(P.Wdg, wb + 1 * HH, HH);
    cvt_f32_to_bf16_kernel<<<nblk(HH, 256), 256, 0, stream>>>(P.Wdu, wb + 2 * HH, HH);
    cvt_f32_to_bf16_kernel<<<nblk(HH, 256), 256, 0, stream>>>(P.Wsu, wb + 3 * HH, HH);
    cvt_f32_to_bf16_kernel<<<nblk(HH, 256), 256, 0, stream>>>(P.Weg, wb + 4 * HH, HH);

    // fused node projections: sg, dg, du, su (su lands directly in `out`)
    gemm128x4_bf16_wmma_kernel<<<(unsigned)(V / 16), 256, 0, stream>>>(
        xb, wb, P.bsg, P.bdg, P.bdu, P.bsu, sg, dg, du, out, (int)V);
    // edge projection
    gemm128_bf16_wmma_kernel<<<(unsigned)(E / 16), 256, 0, stream>>>(
        eab, wb + 4 * HH, P.beg, eglin, (int)E);

    // segment sums
    fill_f32_kernel<<<nblk(VH, 256), 256, 0, stream>>>(norm, 1e-8f, VH);
    edge_norm_kernel<<<nblk(EH, 256), 256, 0, stream>>>(src, dst, sg, dg, eglin, norm, EH);
    edge_msg_kernel<<<nblk(EH, 256), 256, 0, stream>>>(src, dst, sg, dg, eglin, du, norm, out, EH);

    // batch norm (training-mode stats) + silu + residual
    fill_f32_kernel<<<1, 256, 0, stream>>>(stats, 0.f, 256);
    bn_stats_kernel<<<1024, 256, 0, stream>>>(out, stats, V);
    bn_finalize_kernel<<<1, 128, 0, stream>>>(stats, P.gamma, P.beta, (float)V);
    bn_apply_kernel<<<nblk(VH, 256), 256, 0, stream>>>(out, x, stats, out_bf16, VH);
}

static ConvParams load_params(void* const* d_in, int base) {
    ConvParams p;
    p.Wsg = (const float*)d_in[base + 0];  p.bsg = (const float*)d_in[base + 1];
    p.Wdg = (const float*)d_in[base + 2];  p.bdg = (const float*)d_in[base + 3];
    p.Weg = (const float*)d_in[base + 4];  p.beg = (const float*)d_in[base + 5];
    p.Wsu = (const float*)d_in[base + 6];  p.bsu = (const float*)d_in[base + 7];
    p.Wdu = (const float*)d_in[base + 8];  p.bdu = (const float*)d_in[base + 9];
    p.gamma = (const float*)d_in[base + 10];
    p.beta  = (const float*)d_in[base + 11];
    return p;
}

extern "C" void kernel_launch(void* const* d_in, const int* in_sizes, int n_in,
                              void* d_out, int out_size, void* d_ws, size_t ws_size,
                              hipStream_t stream) {
    (void)in_sizes; (void)n_in; (void)out_size; (void)ws_size;

    const float* x_g   = (const float*)d_in[0];
    const int*   ei_g  = (const int*)d_in[1];
    const float* x_lg  = (const float*)d_in[2];
    const int*   ei_lg = (const int*)d_in[3];
    const float* ea_lg = (const float*)d_in[4];
    const ConvParams Plg = load_params(d_in, 5);
    const ConvParams Pg  = load_params(d_in, 17);

    // workspace bump allocator (256B aligned)
    char* w = (char*)d_ws;
    auto take = [&](size_t bytes) -> char* {
        char* p = w;
        w += (bytes + 255) & ~(size_t)255;
        return p;
    };
    const long long HH = (long long)H * H;
    __bf16* wb    = (__bf16*)take(5 * HH * sizeof(__bf16));
    __bf16* xb    = (__bf16*)take((size_t)EGn * H * sizeof(__bf16));   // max V
    __bf16* eab   = (__bf16*)take((size_t)ELGn * H * sizeof(__bf16));  // max E
    float*  sg    = (float*)take((size_t)EGn * H * sizeof(float));
    float*  dg    = (float*)take((size_t)EGn * H * sizeof(float));
    float*  du    = (float*)take((size_t)EGn * H * sizeof(float));
    float*  eglin = (float*)take((size_t)ELGn * H * sizeof(float));
    float*  norm  = (float*)take((size_t)EGn * H * sizeof(float));
    float*  stats = (float*)take(256 * sizeof(float));

    float* out_node = (float*)d_out;                    // x         [N,H]
    float* out_edge = (float*)d_out + (size_t)Nn * H;   // edge_attr [EG,H]

    // --- conv 1: line graph (V = EG, E = ELG) ---
    cvt_f32_to_bf16_kernel<<<nblk(EGn * H, 256), 256, 0, stream>>>(x_lg, xb, EGn * H);
    cvt_f32_to_bf16_kernel<<<nblk(ELGn * H, 256), 256, 0, stream>>>(ea_lg, eab, ELGn * H);
    run_conv(stream, x_lg, EGn, ei_lg, ei_lg + ELGn, ELGn,
             xb, eab, wb, sg, dg, du, eglin, norm, stats, Plg,
             out_edge, /*bf16 copy feeds conv2's edge GEMM*/ eab);

    // --- conv 2: atom graph (V = N, E = EG), edge_attr = conv1 output ---
    cvt_f32_to_bf16_kernel<<<nblk(Nn * H, 256), 256, 0, stream>>>(x_g, xb, Nn * H);
    run_conv(stream, x_g, Nn, ei_g, ei_g + EGn, EGn,
             xb, eab, wb, sg, dg, du, eglin, norm, stats, Pg,
             out_node, nullptr);
}
